// DKT_23656679866371
// MI455X (gfx1250) — compile-verified, matrix operand order checked by reading, not run on hardware
//
#include <hip/hip_runtime.h>

typedef __attribute__((ext_vector_type(16))) __bf16 v16bf;
typedef __attribute__((ext_vector_type(8)))  float  v8f;

#define NB 64
#define LL 500
#define EMB 256
#define NNODE 101
#define TOK (NB*LL)
#define OUTP (NB*(LL-1))

union Frag { uint4 q[2]; v16bf v; };
union AccU { v8f v; float f[8]; };

__device__ inline unsigned short f2bf(float x) {
  union { float f; unsigned u; } c; c.f = x;
  unsigned u = c.u;
  unsigned r = (u + 0x7FFFu + ((u >> 16) & 1u)) >> 16;
  return (unsigned short)r;
}

__device__ inline v8f wmma_bf(const Frag& a, const Frag& b, v8f c) {
  return __builtin_amdgcn_wmma_f32_16x16x32_bf16(false, a.v, false, b.v,
                                                 (short)0, c, false, false);
}

// ---------------- generic f32 -> bf16 convert ----------------
__global__ void k_cvt(const float* __restrict__ in, unsigned short* __restrict__ out, int n) {
  int i = blockIdx.x * 256 + threadIdx.x;
  if (i < n) out[i] = f2bf(in[i]);
}

__global__ void k_addvec(const float* a, const float* b, float* o, int n) {
  int i = threadIdx.x; if (i < n) o[i] = a[i] + b[i];
}

// ---------------- WMMA GEMM: C[M,N] = A[M,K] @ Bw[N,K]^T (+bias) ----------------
// B operand staged to LDS with CDNA5 async global->LDS copies (ASYNCcnt),
// double-buffered so the DMA of chunk k+1 overlaps the WMMAs of chunk k.
// flags: 1 = relu, 2 = accumulate into C, 4 = Cb stored transposed (Cb[n*ldcb+m])
__global__ __launch_bounds__(128) void k_gemm(
    const unsigned short* __restrict__ A, int lda,
    const unsigned short* __restrict__ Bw, int ldb,
    float* __restrict__ C, int ldc,
    unsigned short* __restrict__ Cb, int ldcb,
    const float* __restrict__ bias, int K, int flags) {
  extern __shared__ unsigned short bsm[];   // 2 buffers x (32 rows x 64 bf16) = 8KB
  int lane = threadIdx.x & 31;
  int wave = threadIdx.x >> 5;
  int n0 = blockIdx.x * 32;
  int m0 = blockIdx.y * 64 + wave * 16;
  int mr = lane & 15, hf = lane >> 4;

  const unsigned short* Ar = A + (size_t)(m0 + mr) * lda + hf * 8;

  // async-stage one 32(N) x 64(K) bf16 B tile into LDS buffer `buf`
  auto stage = [&](int kc, int buf) {
    #pragma unroll
    for (int j = 0; j < 2; j++) {
      int t = threadIdx.x + j * 128;          // 256 x 16B transfers = 4KB
      int nl = t >> 3, sub = t & 7;
      unsigned lds = (unsigned)(buf * 4096 + nl * 128 + sub * 16);
      unsigned long long ga = (unsigned long long)(const void*)
          (Bw + (size_t)(n0 + nl) * ldb + kc + sub * 8);
      asm volatile("global_load_async_to_lds_b128 %0, %1, off"
                   :: "v"(lds), "v"(ga) : "memory");
    }
  };

  AccU a0, a1;
  if (flags & 2) {
    #pragma unroll
    for (int j = 0; j < 8; j++) {
      int m = m0 + hf * 8 + j, n = n0 + mr;
      a0.f[j] = C[(size_t)m * ldc + n];
      a1.f[j] = C[(size_t)m * ldc + n + 16];
    }
  } else {
    #pragma unroll
    for (int j = 0; j < 8; j++) { a0.f[j] = 0.f; a1.f[j] = 0.f; }
  }

  stage(0, 0);
  asm volatile("s_wait_asynccnt 0" ::: "memory");
  __syncthreads();

  for (int kc = 0; kc < K; kc += 64) {
    int cur = (kc >> 6) & 1;
    if (kc + 64 < K) stage(kc + 64, cur ^ 1);
    if (kc + 128 < K) __builtin_prefetch(Ar + kc + 128, 0, 0);
    const unsigned short* Bl = bsm + cur * 2048;  // elements
    #pragma unroll
    for (int ks = 0; ks < 2; ks++) {
      Frag a, b0, b1;
      a.q[0] = *(const uint4*)(Ar + kc + ks * 32);
      a.q[1] = *(const uint4*)(Ar + kc + ks * 32 + 16);
      const unsigned short* br0 = Bl + (size_t)mr * 64 + ks * 32 + hf * 8;
      const unsigned short* br1 = Bl + (size_t)(16 + mr) * 64 + ks * 32 + hf * 8;
      b0.q[0] = *(const uint4*)(br0);
      b0.q[1] = *(const uint4*)(br0 + 16);
      b1.q[0] = *(const uint4*)(br1);
      b1.q[1] = *(const uint4*)(br1 + 16);
      a0.v = wmma_bf(a, b0, a0.v);
      a1.v = wmma_bf(a, b1, a1.v);
    }
    if (kc + 64 < K) {
      asm volatile("s_wait_asynccnt 0" ::: "memory");
      __syncthreads();
    }
  }

  #pragma unroll
  for (int j = 0; j < 8; j++) {
    int m = m0 + hf * 8 + j, n = n0 + mr;
    float v0 = a0.f[j] + (bias ? bias[n] : 0.f);
    float v1 = a1.f[j] + (bias ? bias[n + 16] : 0.f);
    if (flags & 1) { v0 = v0 > 0.f ? v0 : 0.f; v1 = v1 > 0.f ? v1 : 0.f; }
    C[(size_t)m * ldc + n]      = v0;
    C[(size_t)m * ldc + n + 16] = v1;
    if (Cb) {
      if (flags & 4) {
        Cb[(size_t)n * ldcb + m]        = f2bf(v0);
        Cb[(size_t)(n + 16) * ldcb + m] = f2bf(v1);
      } else {
        Cb[(size_t)m * ldcb + n]      = f2bf(v0);
        Cb[(size_t)m * ldcb + n + 16] = f2bf(v1);
      }
    }
  }
}

// ---------------- GCN ----------------
__global__ void k_gcn_deg(const int* __restrict__ skill, float* __restrict__ dinv) {
  __shared__ float deg[NNODE];
  int b = blockIdx.x;
  for (int i = threadIdx.x; i < NNODE; i += blockDim.x) deg[i] = 1.0f; // self loop
  __syncthreads();
  for (int e = threadIdx.x; e < LL - 1; e += blockDim.x)
    atomicAdd(&deg[skill[b * LL + e + 1]], 1.0f);
  __syncthreads();
  for (int i = threadIdx.x; i < NNODE; i += blockDim.x)
    dinv[b * NNODE + i] = rsqrtf(deg[i]);
}

__global__ void k_gcn_mm1(const float* __restrict__ se, const float* __restrict__ w1,
                          float* __restrict__ hh1) {
  for (int idx = threadIdx.x; idx < NNODE * 8; idx += blockDim.x) {
    int i = idx / 8, j = idx % 8;
    float s = 0.f;
    for (int k = 0; k < EMB; k++) s += se[i * EMB + k] * w1[j * EMB + k];
    hh1[idx] = s;
  }
}

__global__ void k_gcn_sc1(const int* __restrict__ skill, const float* __restrict__ dinv,
                          const float* __restrict__ hh1, const float* __restrict__ b1,
                          float* __restrict__ r1) {
  __shared__ float out[NNODE * 8];
  __shared__ float dl[NNODE];
  int b = blockIdx.x;
  for (int i = threadIdx.x; i < NNODE; i += blockDim.x) dl[i] = dinv[b * NNODE + i];
  __syncthreads();
  for (int i = threadIdx.x; i < NNODE * 8; i += blockDim.x)
    out[i] = dl[i / 8] * dl[i / 8] * hh1[i];
  __syncthreads();
  for (int e = threadIdx.x / 8; e < LL - 1; e += blockDim.x / 8) {
    int j = threadIdx.x % 8;
    int s = skill[b * LL + e], d = skill[b * LL + e + 1];
    atomicAdd(&out[d * 8 + j], dl[s] * dl[d] * hh1[s * 8 + j]);
  }
  __syncthreads();
  for (int i = threadIdx.x; i < NNODE * 8; i += blockDim.x) {
    float v = out[i] + b1[i % 8];
    r1[b * NNODE * 8 + i] = v > 0.f ? v : 0.f;
  }
}

__global__ void k_gcn_sc2(const int* __restrict__ skill, const float* __restrict__ dinv,
                          const float* __restrict__ r1, const float* __restrict__ w2,
                          const float* __restrict__ b2, float* __restrict__ allh) {
  __shared__ float rl[NNODE * 8];
  __shared__ float dl[NNODE];
  __shared__ float hh[NNODE * 64];
  __shared__ float out[NNODE * 64];
  int b = blockIdx.x;
  for (int i = threadIdx.x; i < NNODE * 8; i += blockDim.x) rl[i] = r1[b * NNODE * 8 + i];
  for (int i = threadIdx.x; i < NNODE; i += blockDim.x) dl[i] = dinv[b * NNODE + i];
  __syncthreads();
  for (int c0 = 0; c0 < EMB; c0 += 64) {
    for (int i = threadIdx.x; i < NNODE * 64; i += blockDim.x) {
      int node = i / 64, cc = i % 64;
      const float* wr = &w2[(c0 + cc) * 8];
      float s = 0.f;
      #pragma unroll
      for (int j = 0; j < 8; j++) s += rl[node * 8 + j] * wr[j];
      hh[i] = s;
      out[i] = dl[node] * dl[node] * s;
    }
    __syncthreads();
    for (int e = threadIdx.x / 64; e < LL - 1; e += blockDim.x / 64) {
      int cc = threadIdx.x % 64;
      int s = skill[b * LL + e], d = skill[b * LL + e + 1];
      atomicAdd(&out[d * 64 + cc], dl[s] * dl[d] * hh[s * 64 + cc]);
    }
    __syncthreads();
    for (int i = threadIdx.x; i < NNODE * 64; i += blockDim.x) {
      int node = i / 64, cc = i % 64;
      allh[((size_t)b * NNODE + node) * EMB + c0 + cc] = out[i] + b2[c0 + cc];
    }
    __syncthreads();
  }
}

// ---------------- attention pooling ----------------
__global__ void k_pool(const int* __restrict__ answer, const int* __restrict__ skill,
                       const float* __restrict__ pos, const float* __restrict__ allh,
                       float* __restrict__ pooled) {
  __shared__ float wm[LL];
  __shared__ int sk[LL];
  __shared__ int lens;
  __shared__ float ssum;
  int b = blockIdx.x;
  if (threadIdx.x == 0) { lens = 0; ssum = 0.f; }
  __syncthreads();
  int cnt = 0;
  for (int l = threadIdx.x; l < LL; l += blockDim.x) {
    sk[l] = skill[b * LL + l];
    cnt += (answer[b * LL + l] != 2);
  }
  atomicAdd(&lens, cnt);
  __syncthreads();
  int row = lens - 1; if (row < 0) row = 0;
  float ls = 0.f;
  for (int l = threadIdx.x; l < LL; l += blockDim.x) {
    float m = (answer[b * LL + l] != 2) ? 1.f : 0.f;
    float e = __expf(pos[(size_t)row * LL + l] * m);
    wm[l] = e * m;  // mask folded in (g already masked in reference)
    ls += e;        // softmax denominator includes masked terms (exp(0)=1)
  }
  atomicAdd(&ssum, ls);
  __syncthreads();
  float inv = 1.f / ssum;
  for (int c = threadIdx.x; c < EMB; c += blockDim.x) {
    float s = 0.f;
    for (int l = 0; l < LL; l++)
      s += allh[((size_t)b * NNODE + sk[l]) * EMB + c] * wm[l];
    pooled[b * EMB + c] = s * inv;
  }
}

// ---------------- RNN input assembly (bf16) ----------------
__global__ void k_xin_dg(const int* __restrict__ skill, const int* __restrict__ answer,
                         const float* __restrict__ se, const float* __restrict__ ae,
                         const float* __restrict__ pooled, unsigned short* __restrict__ xin) {
  int tok = blockIdx.x, b = tok / LL;
  int s = skill[tok], a = answer[tok];
  const float* sp = &se[s * EMB];
  const float* ap = &ae[a * EMB];
  const float* first = (a == 1) ? sp : ap;
  const float* second = (a == 1) ? ap : sp;
  unsigned short* o = &xin[(size_t)tok * (3 * EMB)];
  int t = threadIdx.x;
  o[t]           = f2bf(pooled[b * EMB + t]);
  o[EMB + t]     = f2bf(first[t]);
  o[2 * EMB + t] = f2bf(second[t]);
}

__global__ void k_xin_hgr(const int* __restrict__ student, const int* __restrict__ skill,
                          const int* __restrict__ answer, const unsigned short* __restrict__ stuemb_b,
                          const float* __restrict__ se, const float* __restrict__ ae,
                          unsigned short* __restrict__ xin) {
  int tok = blockIdx.x;
  int s = skill[tok], a = answer[tok];
  const float* sp = &se[s * EMB];
  const float* ap = &ae[a * EMB];
  const float* first = (a == 1) ? sp : ap;
  const float* second = (a == 1) ? ap : sp;
  unsigned short* o = &xin[(size_t)tok * (3 * EMB)];
  int t = threadIdx.x;
  o[t]           = stuemb_b[(size_t)(student[tok] - 1) * EMB + t];
  o[EMB + t]     = f2bf(first[t]);
  o[2 * EMB + t] = f2bf(second[t]);
}

// ---------------- persistent recurrent kernel ----------------
// one workgroup, 32 waves; Whh (128KB) + double-buffered h (64KB) in LDS (CDNA5 320KB WGP)
// h_{t+1} = tanh(xz[:,t,:] + h_t @ Whh^T)
__global__ __launch_bounds__(1024) void k_rnn(
    const unsigned short* __restrict__ Whh_b,   // (256,256) bf16
    const float* __restrict__ xz,               // (64,500,256) f32
    float* __restrict__ hs_f,                   // (64,500,256) f32
    unsigned short* __restrict__ hs_b) {        // (64,500,256) bf16
  extern __shared__ unsigned short smem[];
  unsigned short* whs  = smem;                  // 256*256
  unsigned short* hbuf = smem + EMB * EMB;      // 2 * 64*256

  int lane = threadIdx.x & 31, wave = threadIdx.x >> 5;
  int mtile = wave >> 3, npair = wave & 7;
  int m0 = mtile * 16, n0 = npair * 32;
  int mr = lane & 15, hf = lane >> 4;

  for (int i = threadIdx.x; i < EMB * EMB; i += 1024) whs[i] = Whh_b[i];
  for (int i = threadIdx.x; i < 2 * NB * EMB; i += 1024) hbuf[i] = 0;
  __syncthreads();

  const unsigned short* Brow0 = whs + (size_t)(n0 + mr) * EMB + hf * 8;
  const unsigned short* Brow1 = Brow0 + (size_t)16 * EMB;

  for (int t = 0; t < LL; t++) {
    const unsigned short* hcur = hbuf + (t & 1) * NB * EMB;
    unsigned short* hnx = hbuf + ((t + 1) & 1) * NB * EMB;
    const unsigned short* Ar = hcur + (m0 + mr) * EMB + hf * 8;

    AccU a0, a1;
    #pragma unroll
    for (int j = 0; j < 8; j++) { a0.f[j] = 0.f; a1.f[j] = 0.f; }

    #pragma unroll
    for (int ks = 0; ks < 8; ks++) {
      Frag a, b0, b1;
      a.q[0]  = *(const uint4*)(Ar + ks * 32);
      a.q[1]  = *(const uint4*)(Ar + ks * 32 + 16);
      b0.q[0] = *(const uint4*)(Brow0 + ks * 32);
      b0.q[1] = *(const uint4*)(Brow0 + ks * 32 + 16);
      b1.q[0] = *(const uint4*)(Brow1 + ks * 32);
      b1.q[1] = *(const uint4*)(Brow1 + ks * 32 + 16);
      a0.v = wmma_bf(a, b0, a0.v);
      a1.v = wmma_bf(a, b1, a1.v);
    }

    #pragma unroll
    for (int j = 0; j < 8; j++) {
      int m = m0 + hf * 8 + j, n = n0 + mr;
      size_t g = ((size_t)m * LL + t) * EMB;
      float v0 = tanhf(a0.f[j] + xz[g + n]);
      float v1 = tanhf(a1.f[j] + xz[g + n + 16]);
      hs_f[g + n] = v0; hs_f[g + n + 16] = v1;
      unsigned short c0 = f2bf(v0), c1 = f2bf(v1);
      hs_b[g + n] = c0; hs_b[g + n + 16] = c1;
      hnx[m * EMB + n] = c0; hnx[m * EMB + n + 16] = c1;
    }
    __syncthreads();
  }
}

// ---------------- heads + next_pred (selected-column dots, one wave/token) ----------------
__global__ __launch_bounds__(256) void k_heads(
    const int* __restrict__ skill,
    const float* __restrict__ hs_hgr, const float* __restrict__ hs_dg,
    const float* __restrict__ thp,
    const float* __restrict__ w1_b, const float* __restrict__ w2_b,
    const float* __restrict__ fc_h_w, const float* __restrict__ fc_h_b,
    const float* __restrict__ fc_d_w, const float* __restrict__ fc_d_b,
    const float* __restrict__ fc_e_w, const float* __restrict__ fc_e_b,
    float* __restrict__ out) {
  int gw = (blockIdx.x * blockDim.x + threadIdx.x) >> 5;
  int lane = threadIdx.x & 31;
  if (gw >= OUTP) return;
  int b = gw / (LL - 1), l = gw % (LL - 1);
  int ns = skill[b * LL + l + 1];
  int c = ns < 99 ? ns : 99;
  bool z = (ns == 100);
  size_t base = ((size_t)b * LL + l) * EMB;

  float s_h = 0.f, s_d = 0.f, s_e = 0.f;
  for (int k = lane; k < EMB; k += 32) {
    float hh = hs_hgr[base + k], hd = hs_dg[base + k];
    float th = 1.f / (1.f + __expf(-(thp[base + k] + w1_b[k] + w2_b[k])));
    s_h += hh * fc_h_w[c * EMB + k];
    s_d += hd * fc_d_w[c * EMB + k];
    s_e += th * hh * fc_e_w[c * 2 * EMB + k] + (1.f - th) * hd * fc_e_w[c * 2 * EMB + EMB + k];
  }
  #pragma unroll
  for (int off = 16; off > 0; off >>= 1) {
    s_h += __shfl_down(s_h, off, 32);
    s_d += __shfl_down(s_d, off, 32);
    s_e += __shfl_down(s_e, off, 32);
  }
  if (lane == 0) {
    int o = b * (LL - 1) + l;
    out[o]            = z ? 0.f : (s_h + fc_h_b[c]);
    out[OUTP + o]     = z ? 0.f : (s_d + fc_d_b[c]);
    out[2 * OUTP + o] = z ? 0.f : (s_e + fc_e_b[c]);
  }
}

extern "C" void kernel_launch(void* const* d_in, const int* in_sizes, int n_in,
                              void* d_out, int out_size, void* d_ws, size_t ws_size,
                              hipStream_t stream) {
  const int*   student    = (const int*)d_in[0];
  const int*   skill      = (const int*)d_in[1];
  const int*   answer     = (const int*)d_in[2];
  const float* G          = (const float*)d_in[3];
  const float* skill_emb  = (const float*)d_in[4];
  const float* answer_emb = (const float*)d_in[5];
  const float* stu        = (const float*)d_in[6];
  const float* hg_w1 = (const float*)d_in[7];
  const float* hg_b1 = (const float*)d_in[8];
  const float* hg_w2 = (const float*)d_in[9];
  const float* hg_b2 = (const float*)d_in[10];
  const float* gcn_w1 = (const float*)d_in[11];
  const float* gcn_b1 = (const float*)d_in[12];
  const float* gcn_w2 = (const float*)d_in[13];
  const float* gcn_b2 = (const float*)d_in[14];
  const float* w1_w = (const float*)d_in[15];
  const float* w1_b = (const float*)d_in[16];
  const float* w2_w = (const float*)d_in[17];
  const float* w2_b = (const float*)d_in[18];
  const float* fc_d_w = (const float*)d_in[19];
  const float* fc_d_b = (const float*)d_in[20];
  const float* fc_h_w = (const float*)d_in[21];
  const float* fc_h_b = (const float*)d_in[22];
  const float* fc_e_w = (const float*)d_in[23];
  const float* fc_e_b = (const float*)d_in[24];
  const float* pos = (const float*)d_in[25];
  const float* dg_wih = (const float*)d_in[26];
  const float* dg_whh = (const float*)d_in[27];
  const float* dg_bih = (const float*)d_in[28];
  const float* dg_bhh = (const float*)d_in[29];
  const float* hgr_wih = (const float*)d_in[30];
  const float* hgr_whh = (const float*)d_in[31];
  const float* hgr_bih = (const float*)d_in[32];
  const float* hgr_bhh = (const float*)d_in[33];
  float* out = (float*)d_out;

  // ---- workspace carve ----
  char* w = (char*)d_ws;
  size_t off = 0;
  auto alloc = [&](size_t bytes) -> void* {
    void* p = w + off;
    off = (off + bytes + 255) & ~(size_t)255;
    return p;
  };
  typedef unsigned short u16;
  u16* Gb      = (u16*)alloc((size_t)4096 * 4096 * 2);
  u16* stub    = (u16*)alloc((size_t)4096 * EMB * 2);
  u16* whg1b   = (u16*)alloc((size_t)EMB * EMB * 2);
  u16* whg2b   = (u16*)alloc((size_t)EMB * EMB * 2);
  u16* w1b     = (u16*)alloc((size_t)EMB * EMB * 2);
  u16* w2b     = (u16*)alloc((size_t)EMB * EMB * 2);
  u16* wdgib   = (u16*)alloc((size_t)EMB * 3 * EMB * 2);
  u16* whgib   = (u16*)alloc((size_t)EMB * 3 * EMB * 2);
  u16* wdghb   = (u16*)alloc((size_t)EMB * EMB * 2);
  u16* whghb   = (u16*)alloc((size_t)EMB * EMB * 2);
  u16* t1bT    = (u16*)alloc((size_t)EMB * 4096 * 2);   // transposed (256,4096)
  u16* hb      = (u16*)alloc((size_t)4096 * EMB * 2);
  u16* t2bT    = (u16*)alloc((size_t)EMB * 4096 * 2);
  u16* stuembb = (u16*)alloc((size_t)4096 * EMB * 2);
  float* tf    = (float*)alloc((size_t)4096 * EMB * 4); // f32 C scratch
  float* dinv  = (float*)alloc((size_t)NB * NNODE * 4);
  float* hh1   = (float*)alloc((size_t)NNODE * 8 * 4);
  float* r1    = (float*)alloc((size_t)NB * NNODE * 8 * 4);
  float* allh  = (float*)alloc((size_t)NB * NNODE * EMB * 4);
  float* pooled= (float*)alloc((size_t)NB * EMB * 4);
  float* bsdg  = (float*)alloc((size_t)EMB * 4);
  float* bshg  = (float*)alloc((size_t)EMB * 4);
  u16* xin     = (u16*)alloc((size_t)TOK * 3 * EMB * 2);  // reused for both RNNs
  float* xz_dg = (float*)alloc((size_t)TOK * EMB * 4);
  float* xz_hg = (float*)alloc((size_t)TOK * EMB * 4);
  float* hsd_f = (float*)alloc((size_t)TOK * EMB * 4);
  u16*   hsd_b = (u16*)alloc((size_t)TOK * EMB * 2);
  float* hsh_f = (float*)alloc((size_t)TOK * EMB * 4);
  u16*   hsh_b = (u16*)alloc((size_t)TOK * EMB * 2);
  float* thp   = (float*)alloc((size_t)TOK * EMB * 4);
  (void)ws_size; (void)in_sizes; (void)n_in; (void)out_size;

  auto cvt = [&](const float* src, u16* dst, size_t n) {
    k_cvt<<<dim3((unsigned)((n + 255) / 256)), dim3(256), 0, stream>>>(src, dst, (int)n);
  };
  auto gemm = [&](const u16* A, int lda, const u16* Bw, int ldb, float* C, int ldc,
                  u16* Cb, int ldcb, const float* bias, int M, int N, int K, int flags) {
    k_gemm<<<dim3(N / 32, M / 64), dim3(128), 8192, stream>>>(A, lda, Bw, ldb, C, ldc,
                                                              Cb, ldcb, bias, K, flags);
  };

  // ---- bf16 conversions ----
  cvt(G, Gb, (size_t)4096 * 4096);
  cvt(stu, stub, (size_t)4096 * EMB);
  cvt(hg_w1, whg1b, EMB * EMB);
  cvt(hg_w2, whg2b, EMB * EMB);
  cvt(w1_w, w1b, EMB * EMB);
  cvt(w2_w, w2b, EMB * EMB);
  cvt(dg_wih, wdgib, EMB * 3 * EMB);
  cvt(hgr_wih, whgib, EMB * 3 * EMB);
  cvt(dg_whh, wdghb, EMB * EMB);
  cvt(hgr_whh, whghb, EMB * EMB);

  // ---- hypergraph student embedding: G @ (relu(G @ (stu@hg_w1^T+b1)) @ hg_w2^T + b2) ----
  gemm(stub, EMB, whg1b, EMB, tf, EMB, t1bT, 4096, hg_b1, 4096, EMB, EMB, 4);        // t1 (T)
  gemm(Gb, 4096, t1bT, 4096, tf, EMB, hb, EMB, nullptr, 4096, EMB, 4096, 1);         // relu(G@t1)
  gemm(hb, EMB, whg2b, EMB, tf, EMB, t2bT, 4096, hg_b2, 4096, EMB, EMB, 4);          // t2 (T)
  gemm(Gb, 4096, t2bT, 4096, tf, EMB, stuembb, EMB, nullptr, 4096, EMB, 4096, 0);    // stu_emb

  // ---- 2-layer GCN per batch ----
  k_gcn_deg<<<NB, 256, 0, stream>>>(skill, dinv);
  k_gcn_mm1<<<1, 256, 0, stream>>>(skill_emb, gcn_w1, hh1);
  k_gcn_sc1<<<NB, 256, 0, stream>>>(skill, dinv, hh1, gcn_b1, r1);
  k_gcn_sc2<<<NB, 256, 0, stream>>>(skill, dinv, r1, gcn_w2, gcn_b2, allh);

  // ---- attention pooling ----
  k_pool<<<NB, 256, 0, stream>>>(answer, skill, pos, allh, pooled);

  // ---- RNN input projections (xz = in @ Wih^T + bih + bhh) ----
  k_addvec<<<1, 256, 0, stream>>>(dg_bih, dg_bhh, bsdg, EMB);
  k_addvec<<<1, 256, 0, stream>>>(hgr_bih, hgr_bhh, bshg, EMB);
  k_xin_dg<<<TOK, 256, 0, stream>>>(skill, answer, skill_emb, answer_emb, pooled, xin);
  gemm(xin, 3 * EMB, wdgib, 3 * EMB, xz_dg, EMB, nullptr, 0, bsdg, TOK, EMB, 3 * EMB, 0);
  k_xin_hgr<<<TOK, 256, 0, stream>>>(student, skill, answer, stuembb, skill_emb, answer_emb, xin);
  gemm(xin, 3 * EMB, whgib, 3 * EMB, xz_hg, EMB, nullptr, 0, bshg, TOK, EMB, 3 * EMB, 0);

  // ---- recurrent loops (persistent single-WGP kernels, 196KB LDS each) ----
  size_t smem = (size_t)EMB * EMB * 2 + (size_t)2 * NB * EMB * 2;
  k_rnn<<<1, 1024, smem, stream>>>(wdghb, xz_dg, hsd_f, hsd_b);
  k_rnn<<<1, 1024, smem, stream>>>(whghb, xz_hg, hsh_f, hsh_b);

  // ---- theta pre-activation: thp = h_hg@w1_w^T + h_dg@w2_w^T ----
  gemm(hsh_b, EMB, w1b, EMB, thp, EMB, nullptr, 0, nullptr, TOK, EMB, EMB, 0);
  gemm(hsd_b, EMB, w2b, EMB, thp, EMB, nullptr, 0, nullptr, TOK, EMB, EMB, 2);

  // ---- heads + next_pred ----
  k_heads<<<OUTP / 8, 256, 0, stream>>>(skill, hsh_f, hsd_f, thp, w1_b, w2_b,
                                        fc_h_w, fc_h_b, fc_d_w, fc_d_b,
                                        fc_e_w, fc_e_b, out);
}
